// NodeScalarFeatInit_20907900797502
// MI455X (gfx1250) — compile-verified, matrix operand order checked by reading, not run on hardware
//
#include <hip/hip_runtime.h>
#include <hip/hip_bf16.h>

// ---------------------------------------------------------------------------
// Problem constants (reference: b=4, n=2048, d=256)
// ---------------------------------------------------------------------------
#define BATCH   4
#define NN      2048
#define DD      256
#define DD2     512
#define LN_EPSF 1e-5f

typedef __attribute__((ext_vector_type(16))) _Float16 v16h;
typedef __attribute__((ext_vector_type(8)))  _Float16 v8h;
typedef __attribute__((ext_vector_type(4)))  _Float16 v4h;
typedef __attribute__((ext_vector_type(8)))  float    v8f;

// LDS tile row stride in halves (padded to dodge bank conflicts)
#define LDT 40

__device__ __forceinline__ v8f vzero8f() {
  v8f z;
#pragma unroll
  for (int i = 0; i < 8; ++i) z[i] = 0.0f;
  return z;
}

// Load a 16x32 f16 operand fragment from an LDS tile stored with contiguous
// K (stride `ld` halves per row/column).  Per ISA 7.12.2 (16-bit A matrix):
//   lanes 0-15  : row = lane,    K = 0..7 (v0-3)  and 16..23 (v4-7)
//   lanes 16-31 : row = lane-16, K = 8..15 (v0-3) and 24..31 (v4-7)
// B operand uses the same loader with the tile stored column-major (K along
// the fast axis), i.e. the loader reads column `lane&15` of B.
__device__ __forceinline__ v16h frag_ld(const _Float16* base, int ld) {
  const int lane = threadIdx.x & 31;
  const int r    = lane & 15;
  const int ko   = (lane >> 4) << 3;      // 0 or 8
  const _Float16* p = base + r * ld + ko;
  v8h lo = *(const v8h*)(p);
  v8h hi = *(const v8h*)(p + 16);
  v16h out;
#pragma unroll
  for (int i = 0; i < 8; ++i) { out[i] = lo[i]; out[i + 8] = hi[i]; }
  return out;
}

__device__ __forceinline__ v8f wmma16(v16h a, v16h b, v8f c) {
  // D = A(16x32,f16) * B(32x16,f16) + C(16x16,f32)
  return __builtin_amdgcn_wmma_f32_16x16x32_f16(
      /*neg_a=*/false, a, /*neg_b=*/false, b,
      /*c_mod=*/(short)0, c, /*reuse_a=*/false, /*reuse_b=*/false);
}

// ---------------------------------------------------------------------------
// CDNA5 async global->LDS copy (16 bytes per lane), tracked by ASYNCcnt.
// LDS byte offset lds_off is filled from global address gsrc (per lane).
// ---------------------------------------------------------------------------
__device__ __forceinline__ void async_copy_b128(unsigned lds_off, const void* gsrc) {
  asm volatile("global_load_async_to_lds_b128 %0, %1, off"
               : : "v"(lds_off), "v"(gsrc) : "memory");
}

template <int N>
__device__ __forceinline__ void wait_asynccnt() {
  asm volatile("s_wait_asynccnt %0" : : "i"(N) : "memory");
}

__device__ __forceinline__ unsigned lds_offset_of(const void* p) {
  // Generic pointers to LDS carry the byte offset in the low 32 bits.
  return (unsigned)(unsigned long long)p;
}

// ---------------------------------------------------------------------------
// Kernel 1: convert w1 [512,256] and w2 [256,256] to f16, transposed so the
// B operand has K contiguous:  w1cT[d][k], w2cT[d][k].
// ---------------------------------------------------------------------------
__global__ __launch_bounds__(256)
void prep_weights_kernel(const float* __restrict__ w1, const float* __restrict__ w2,
                         _Float16* __restrict__ w1cT, _Float16* __restrict__ w2cT) {
  const int idx = blockIdx.x * 256 + threadIdx.x;
  if (idx < DD2 * DD) {                    // w1: [k=512][d=256]
    const int k = idx / DD, d = idx % DD;
    w1cT[d * DD2 + k] = (_Float16)w1[idx];
  } else {                                 // w2: [k=256][d=256]
    const int i2 = idx - DD2 * DD;
    const int k = i2 / DD, d = i2 % DD;
    w2cT[d * DD + k] = (_Float16)w2[i2];
  }
}

// ---------------------------------------------------------------------------
// Kernel 2: gather per (b,j):
//   x[b*N+j][0:256]       = f16(atom_embed[id])           (A of MLP1)
//   nbrT[b][k][j]         = f16(neighbor_embed[id][k])    (col-major B of GEMM)
// ---------------------------------------------------------------------------
__global__ __launch_bounds__(256)
void gather_kernel(const int* __restrict__ atom_ids,
                   const float* __restrict__ atom_embed,
                   const float* __restrict__ neighbor_embed,
                   _Float16* __restrict__ xbuf,
                   _Float16* __restrict__ nbrT) {
  const int bj = blockIdx.x;               // 0 .. B*N-1
  const int b  = bj >> 11;
  const int j  = bj & (NN - 1);
  const int k  = threadIdx.x;              // 0..255
  const int id = atom_ids[bj];
  xbuf[(long)bj * DD2 + k] = (_Float16)atom_embed[id * DD + k];
  nbrT[((long)(b * DD + k)) * NN + j] = (_Float16)neighbor_embed[id * DD + k];
}

// ---------------------------------------------------------------------------
// Kernel 3: neighbor_feats = (mask * rel_dist) @ nbr_embeds, scaled by w_rd.
// Per batch GEMM  M=2048, N=256, K=2048.  Block tile 128x128, K-chunk 32.
// 8 waves: waveM in 0..3 (32 rows each), waveN in 0..1 (64 cols each).
// A operand built on the fly (branchless mask), software-pipelined:
//   - A(i+1) global loads prefetched into registers during compute(i)
//   - B staged with async global->LDS (double buffered, ASYNCcnt-fenced)
// Output -> xbuf[:, 256:512] in f16 (A of MLP1).
// ---------------------------------------------------------------------------
__global__ __launch_bounds__(256)
void neighbor_gemm_kernel(const unsigned char* __restrict__ adj,
                          const float* __restrict__ rel,
                          const _Float16* __restrict__ nbrT,
                          const float* __restrict__ w_rd,
                          _Float16* __restrict__ xbuf) {
  __shared__ __align__(16) _Float16 As[128 * LDT];
  __shared__ __align__(16) _Float16 Bs[2][128 * LDT];

  const int t     = threadIdx.x;
  const int wave  = t >> 5;
  const int lane  = t & 31;
  const int bN    = blockIdx.x;            // 0..1   (N/128)
  const int bM    = blockIdx.y;            // 0..15  (M/128)
  const int b     = blockIdx.z;            // batch
  const int waveM = wave >> 1;             // 0..3
  const int waveN = wave & 1;              // 0..1

  v8f acc[2][4];
#pragma unroll
  for (int mt = 0; mt < 2; ++mt)
#pragma unroll
    for (int nt = 0; nt < 4; ++nt) acc[mt][nt] = vzero8f();

  // Staging roles (256 threads): thread -> (row/col = t>>1, 16 K at koStage)
  const int rowStage = t >> 1;
  const int koStage  = (t & 1) * 16;
  const int gi       = bM * 128 + rowStage;
  const long relRow  = ((long)(b * NN + gi)) * NN;
  const _Float16* nbrBase =
      nbrT + ((long)(b * DD + bN * 128 + rowStage)) * NN + koStage;
  const unsigned ldsAoff = rowStage * LDT + koStage;

  float4 pr[4];          // prefetched rel_dist (16 floats)
  int4   pa;             // prefetched adj bytes (16 bools)

  // ---- prologue: prefetch chunk 0 ----
  {
    const float4* rp4 = (const float4*)(rel + relRow + koStage);
#pragma unroll
    for (int q = 0; q < 4; ++q) pr[q] = rp4[q];
    pa = *(const int4*)(adj + relRow + koStage);
  }
  {
    const char* g = (const char*)(nbrBase);
    const unsigned l = lds_offset_of(&Bs[0][ldsAoff]);
    async_copy_b128(l, g);
    async_copy_b128(l + 16, g + 16);
  }

  const int NITER = NN / 32;               // 64
  for (int it = 0; it < NITER; ++it) {
    const int kc = it * 32;
    __syncthreads();   // compute(it-1) done: As free, Bs[(it+1)&1] free

    // ---- store A(it): branchless mask*rel -> f16 ----
    {
      const int av[4] = {pa.x, pa.y, pa.z, pa.w};
      float rf[16];
#pragma unroll
      for (int q = 0; q < 4; ++q) {
        rf[q * 4 + 0] = pr[q].x; rf[q * 4 + 1] = pr[q].y;
        rf[q * 4 + 2] = pr[q].z; rf[q * 4 + 3] = pr[q].w;
      }
#pragma unroll
      for (int e = 0; e < 16; ++e) {
        const int gj = kc + koStage + e;
        const unsigned m = (unsigned)((av[e >> 2] >> ((e & 3) * 8)) & 0xff);
        // keep = (adj != 0) & (i != j); branchless float masking via AND.
        const unsigned keep = (unsigned)(m != 0u) & (unsigned)(gi != gj);
        union { float f; unsigned u; } cv;
        cv.f = rf[e];
        cv.u &= (0u - keep);
        As[ldsAoff + e] = (_Float16)cv.f;
      }
    }

    // ---- prefetch chunk it+1 (A regs + async B), then fence B(it) ----
    if (it + 1 < NITER) {
      const int kn = kc + 32;
      {
        const char* g = (const char*)(nbrBase + kn);
        const unsigned l = lds_offset_of(&Bs[(it + 1) & 1][ldsAoff]);
        async_copy_b128(l, g);
        async_copy_b128(l + 16, g + 16);
      }
      {
        const float4* rp4 = (const float4*)(rel + relRow + kn + koStage);
#pragma unroll
        for (int q = 0; q < 4; ++q) pr[q] = rp4[q];
        pa = *(const int4*)(adj + relRow + kn + koStage);
      }
      wait_asynccnt<2>();   // the 2 just-issued may be in flight; B(it) done
    } else {
      wait_asynccnt<0>();
    }
    __syncthreads();        // A(it) + B(it) visible to all waves

    // ---- compute: 2 A frags x 4 B frags -> 8 WMMAs ----
    const _Float16* bbuf = &Bs[it & 1][0];
    v16h af[2], bf[4];
#pragma unroll
    for (int mt = 0; mt < 2; ++mt)
      af[mt] = frag_ld(&As[(waveM * 32 + mt * 16) * LDT], LDT);
#pragma unroll
    for (int nt = 0; nt < 4; ++nt)
      bf[nt] = frag_ld(bbuf + (waveN * 64 + nt * 16) * LDT, LDT);
#pragma unroll
    for (int mt = 0; mt < 2; ++mt)
#pragma unroll
      for (int nt = 0; nt < 4; ++nt)
        acc[mt][nt] = wmma16(af[mt], bf[nt], acc[mt][nt]);
  }

  // ---- epilogue: scale by w_rd, write f16 into x[:, 256:512] ----
  const int laneC = lane & 15;
  const int laneR = (lane >> 4) << 3;
  const int cBase = bN * 128 + waveN * 64;
  const int rBase = bM * 128 + waveM * 32;
#pragma unroll
  for (int nt = 0; nt < 4; ++nt) {
    const int col = cBase + nt * 16 + laneC;
    const float wr = w_rd[col];
#pragma unroll
    for (int mt = 0; mt < 2; ++mt) {
      const int row0 = rBase + mt * 16 + laneR;
#pragma unroll
      for (int r = 0; r < 8; ++r) {
        const long row = b * NN + row0 + r;
        xbuf[row * DD2 + DD + col] = (_Float16)(acc[mt][nt][r] * wr);
      }
    }
  }
}

// ---------------------------------------------------------------------------
// Kernel 4: h = x @ w1 + b1 ; LayerNorm ; SiLU ; -> hbuf (f16).
// M = 8192 rows total, block owns 32 rows x all 256 cols (LN needs full row).
// K = 512 in chunks of 32.  8 waves, each owns 32 cols (2x2 WMMA tiles).
// ---------------------------------------------------------------------------
__global__ __launch_bounds__(256)
void mlp1_ln_silu_kernel(const _Float16* __restrict__ xbuf,
                         const _Float16* __restrict__ w1cT,
                         const float* __restrict__ b1,
                         const float* __restrict__ gamma,
                         const float* __restrict__ beta,
                         _Float16* __restrict__ hbuf) {
  __shared__ __align__(16) _Float16 As[32 * LDT];
  __shared__ __align__(16) _Float16 Bs[256 * LDT];
  __shared__ float Hs[32][DD];
  __shared__ float Psum[32][8], Psq[32][8];
  __shared__ float Mu[32], Rstd[32];

  const int t    = threadIdx.x;
  const int wave = t >> 5;
  const int lane = t & 31;
  const long rowBase = (long)blockIdx.x * 32;

  v8f acc[2][2];
#pragma unroll
  for (int mt = 0; mt < 2; ++mt)
#pragma unroll
    for (int nt = 0; nt < 2; ++nt) acc[mt][nt] = vzero8f();

  for (int kc = 0; kc < DD2; kc += 32) {
    __syncthreads();
    {   // stage A: 32 rows x 32 K from xbuf (row-major)
      const int i = t >> 3, ko = (t & 7) * 4;
      *(v4h*)&As[i * LDT + ko] =
          *(const v4h*)(xbuf + (rowBase + i) * DD2 + kc + ko);
    }
    {   // stage B: 256 cols x 32 K from w1cT (col-major, K contiguous)
      const v8h* sp = (const v8h*)(w1cT + (long)t * DD2 + kc);
#pragma unroll
      for (int q = 0; q < 4; ++q) *(v8h*)&Bs[t * LDT + q * 8] = sp[q];
    }
    __syncthreads();
    v16h a0 = frag_ld(&As[0], LDT);
    v16h a1 = frag_ld(&As[16 * LDT], LDT);
    v16h b0 = frag_ld(&Bs[(wave * 32) * LDT], LDT);
    v16h b1f = frag_ld(&Bs[(wave * 32 + 16) * LDT], LDT);
    acc[0][0] = wmma16(a0, b0,  acc[0][0]);
    acc[0][1] = wmma16(a0, b1f, acc[0][1]);
    acc[1][0] = wmma16(a1, b0,  acc[1][0]);
    acc[1][1] = wmma16(a1, b1f, acc[1][1]);
  }

  // ---- h (+bias) into LDS for LayerNorm ----
  __syncthreads();
  const int laneC = lane & 15;
  const int laneR = (lane >> 4) << 3;
#pragma unroll
  for (int nt = 0; nt < 2; ++nt) {
    const int col = wave * 32 + nt * 16 + laneC;
    const float bb = b1[col];
#pragma unroll
    for (int mt = 0; mt < 2; ++mt)
#pragma unroll
      for (int r = 0; r < 8; ++r)
        Hs[mt * 16 + laneR + r][col] = acc[mt][nt][r] + bb;
  }
  __syncthreads();

  // ---- mean / variance (two-level reduction) ----
  {
    const int r = t >> 3, s = t & 7;
    float sum = 0.0f, sq = 0.0f;
#pragma unroll 8
    for (int c = s * 32; c < s * 32 + 32; ++c) {
      const float v = Hs[r][c];
      sum += v; sq += v * v;
    }
    Psum[r][s] = sum; Psq[r][s] = sq;
  }
  __syncthreads();
  if (t < 32) {
    float s = 0.0f, q = 0.0f;
#pragma unroll
    for (int i = 0; i < 8; ++i) { s += Psum[t][i]; q += Psq[t][i]; }
    const float mu = s * (1.0f / DD);
    const float var = q * (1.0f / DD) - mu * mu;
    Mu[t] = mu;
    Rstd[t] = rsqrtf(var + LN_EPSF);
  }
  __syncthreads();

  // ---- normalize, affine, SiLU, store f16 ----
  {
    const int r = t >> 3, s = t & 7;
    const float mu = Mu[r], rs = Rstd[r];
    _Float16* dst = hbuf + (rowBase + r) * DD;
#pragma unroll 8
    for (int c = s * 32; c < s * 32 + 32; ++c) {
      const float v = (Hs[r][c] - mu) * rs * gamma[c] + beta[c];
      const float o = v / (1.0f + __expf(-v));   // silu
      dst[c] = (_Float16)o;
    }
  }
}

// ---------------------------------------------------------------------------
// Kernel 5: out = h @ w2 + b2  (fp32 output).
// M = 8192, N = 256, K = 256.  Block tile 64x256; 8 waves as 2x4
// (32 rows x 64 cols each, 2x4 WMMA tiles).
// ---------------------------------------------------------------------------
__global__ __launch_bounds__(256)
void mlp2_kernel(const _Float16* __restrict__ hbuf,
                 const _Float16* __restrict__ w2cT,
                 const float* __restrict__ b2,
                 float* __restrict__ out) {
  __shared__ __align__(16) _Float16 As[64 * LDT];
  __shared__ __align__(16) _Float16 Bs[256 * LDT];

  const int t     = threadIdx.x;
  const int wave  = t >> 5;
  const int lane  = t & 31;
  const int waveM = wave >> 2;             // 0..1
  const int waveN = wave & 3;              // 0..3
  const long rowBase = (long)blockIdx.x * 64;

  v8f acc[2][4];
#pragma unroll
  for (int mt = 0; mt < 2; ++mt)
#pragma unroll
    for (int nt = 0; nt < 4; ++nt) acc[mt][nt] = vzero8f();

  for (int kc = 0; kc < DD; kc += 32) {
    __syncthreads();
    {   // stage A: 64 rows x 32 K from hbuf
      const int i = t >> 2, ko = (t & 3) * 8;
      *(v8h*)&As[i * LDT + ko] =
          *(const v8h*)(hbuf + (rowBase + i) * DD + kc + ko);
    }
    {   // stage B: 256 cols x 32 K from w2cT
      const v8h* sp = (const v8h*)(w2cT + (long)t * DD + kc);
#pragma unroll
      for (int q = 0; q < 4; ++q) *(v8h*)&Bs[t * LDT + q * 8] = sp[q];
    }
    __syncthreads();
    v16h af[2], bf[4];
#pragma unroll
    for (int mt = 0; mt < 2; ++mt)
      af[mt] = frag_ld(&As[(waveM * 32 + mt * 16) * LDT], LDT);
#pragma unroll
    for (int nt = 0; nt < 4; ++nt)
      bf[nt] = frag_ld(&Bs[(waveN * 64 + nt * 16) * LDT], LDT);
#pragma unroll
    for (int mt = 0; mt < 2; ++mt)
#pragma unroll
      for (int nt = 0; nt < 4; ++nt)
        acc[mt][nt] = wmma16(af[mt], bf[nt], acc[mt][nt]);
  }

  const int laneC = lane & 15;
  const int laneR = (lane >> 4) << 3;
#pragma unroll
  for (int nt = 0; nt < 4; ++nt) {
    const int col = waveN * 64 + nt * 16 + laneC;
    const float bb = b2[col];
#pragma unroll
    for (int mt = 0; mt < 2; ++mt) {
      const long row0 = rowBase + waveM * 32 + mt * 16 + laneR;
#pragma unroll
      for (int r = 0; r < 8; ++r)
        out[(row0 + r) * DD + col] = acc[mt][nt][r] + bb;
    }
  }
}

// ---------------------------------------------------------------------------
// Launcher
// ---------------------------------------------------------------------------
extern "C" void kernel_launch(void* const* d_in, const int* in_sizes, int n_in,
                              void* d_out, int out_size, void* d_ws, size_t ws_size,
                              hipStream_t stream) {
  const int*           atom_ids       = (const int*)d_in[0];
  const unsigned char* adj            = (const unsigned char*)d_in[1];  // bool
  const float*         rel            = (const float*)d_in[2];
  const float*         atom_embed     = (const float*)d_in[3];
  const float*         neighbor_embed = (const float*)d_in[4];
  const float*         w_rd           = (const float*)d_in[5];
  const float*         w1             = (const float*)d_in[6];
  const float*         b1             = (const float*)d_in[7];
  const float*         gamma          = (const float*)d_in[8];
  const float*         beta           = (const float*)d_in[9];
  const float*         w2             = (const float*)d_in[10];
  const float*         b2             = (const float*)d_in[11];
  float* out = (float*)d_out;

  // Workspace carve-up (all f16):
  char* ws = (char*)d_ws;
  _Float16* nbrT = (_Float16*)ws; ws += (size_t)BATCH * DD * NN * 2;   // 4 MB
  _Float16* xbuf = (_Float16*)ws; ws += (size_t)BATCH * NN * DD2 * 2;  // 8 MB
  _Float16* hbuf = (_Float16*)ws; ws += (size_t)BATCH * NN * DD * 2;   // 4 MB
  _Float16* w1cT = (_Float16*)ws; ws += (size_t)DD2 * DD * 2;          // 256 KB
  _Float16* w2cT = (_Float16*)ws; ws += (size_t)DD * DD * 2;           // 128 KB

  prep_weights_kernel<<<(DD2 * DD + DD * DD) / 256, 256, 0, stream>>>(
      w1, w2, w1cT, w2cT);

  gather_kernel<<<BATCH * NN, 256, 0, stream>>>(
      atom_ids, atom_embed, neighbor_embed, xbuf, nbrT);

  dim3 g3(DD / 128, NN / 128, BATCH);     // (2, 16, 4)
  neighbor_gemm_kernel<<<g3, 256, 0, stream>>>(adj, rel, nbrT, w_rd, xbuf);

  mlp1_ln_silu_kernel<<<(BATCH * NN) / 32, 256, 0, stream>>>(
      xbuf, w1cT, b1, gamma, beta, hbuf);

  mlp2_kernel<<<(BATCH * NN) / 64, 256, 0, stream>>>(hbuf, w2cT, b2, out);
}
